// SNNLoss_36782099923663
// MI455X (gfx1250) — compile-verified, matrix operand order checked by reading, not run on hardware
//
#include <hip/hip_runtime.h>

// ---------------------------------------------------------------------------
// SNN loss, fused for CDNA5 (gfx1250, wave32, WMMA + TDM).
//   sim = (x/||x||)(x/||x||)^T / clamp(temp,0.1,1)  [8192x8192, never stored]
//   loss_i = log(tot_i) - log(pos_i);  final = class-weighted scalar.
// Compute bound: 2*N^2*D = 17.2 GFLOP on f16 WMMA, 67M irreducible v_exp_f32.
// Column space split 4x (partial tot/pos per row, summed deterministically)
// to lift occupancy from ~1 to ~5 waves/SIMD. B tiles are TDM-DMA'd (async,
// TENSORcnt, 4-deep pipeline) into LDS with DMA-inserted row padding
// (256B -> 272B stride) so fragment ds_load_b128 are bank-conflict-free.
// ---------------------------------------------------------------------------

typedef __attribute__((ext_vector_type(16))) _Float16     v16h;
typedef __attribute__((ext_vector_type(8)))  _Float16     h8;
typedef __attribute__((ext_vector_type(8)))  float        v8f;
typedef __attribute__((ext_vector_type(4)))  float        f4;
typedef __attribute__((ext_vector_type(4)))  _Float16     h4;
typedef __attribute__((ext_vector_type(4)))  unsigned int u32x4;
typedef __attribute__((ext_vector_type(8)))  int          i32x8;
typedef __attribute__((ext_vector_type(4)))  int          i32x4;

#define SNN_N     8192
#define SNN_D     128
#define SNN_NCLS  16
#define SNN_EPS   1e-8f
#define SNN_LOG2E 1.44269504088896340736f
#define SNN_LN2   0.69314718055994530942f

#if __has_builtin(__builtin_amdgcn_tensor_load_to_lds)
#define SNN_USE_TDM 1
#else
#define SNN_USE_TDM 0
#endif

#define SNN_NBUF  4       // TDM pipeline depth (in-order completion)
#define SNN_ROWB  272     // padded LDS row stride: 256B data + 16B pad
                          // (68 banks -> rows shift 4 banks; no conflicts)
#define SNN_SPLIT 4       // column-space split for occupancy
#define SNN_JTPER (SNN_N / 16 / SNN_SPLIT)   // 128 j-tiles per block

__device__ __forceinline__ float fast_exp2(float x) {
#if __has_builtin(__builtin_amdgcn_exp2f)
  return __builtin_amdgcn_exp2f(x);          // v_exp_f32
#else
  return exp2f(x);
#endif
}
__device__ __forceinline__ float fast_log2(float x) {
#if __has_builtin(__builtin_amdgcn_logf)
  return __builtin_amdgcn_logf(x);           // v_log_f32
#else
  return log2f(x);
#endif
}

// ---------------------------------------------------------------------------
// Kernel 1: per-row L2 normalize f32 -> f16. One wave32 per row.
// ---------------------------------------------------------------------------
__global__ void __launch_bounds__(256) snn_normalize(const float* __restrict__ in,
                                                     _Float16* __restrict__ xn) {
  const int wave = threadIdx.x >> 5;
  const int lane = threadIdx.x & 31;
  const int row  = blockIdx.x * 8 + wave;

  const f4 x = ((const f4*)(in + (size_t)row * SNN_D))[lane];
  float ss = x.x * x.x + x.y * x.y + x.z * x.z + x.w * x.w;
#pragma unroll
  for (int m = 1; m < 32; m <<= 1) ss += __shfl_xor(ss, m, 32);

  const float scale = 1.0f / fmaxf(sqrtf(ss), SNN_EPS);
  h4 o;
  o.x = (_Float16)(x.x * scale);
  o.y = (_Float16)(x.y * scale);
  o.z = (_Float16)(x.z * scale);
  o.w = (_Float16)(x.w * scale);
  ((h4*)(xn + (size_t)row * SNN_D))[lane] = o;
}

// ---------------------------------------------------------------------------
// Fragment loader for V_WMMA_F32_16X16X32_F16 (ISA 16-bit A/B layout):
// lane L -> row L%16; halves e<8 -> K = e + 8*(L/16); e>=8 -> K = e+8+8*(L/16).
// ---------------------------------------------------------------------------
__device__ __forceinline__ v16h snn_frag(const _Float16* rowp, int kk, int h) {
  const _Float16* p = rowp + kk * 32 + 8 * h;          // 16B aligned
  h8 lo = *(const h8*)(p);
  h8 hi = *(const h8*)(p + 16);
  return __builtin_shufflevector(lo, hi, 0, 1, 2, 3, 4, 5, 6, 7,
                                 8, 9, 10, 11, 12, 13, 14, 15);
}

// ---------------------------------------------------------------------------
// TDM: async-copy one 16-row x 256B tile (4 KB) of xn into LDS, inserting
// 16B of LDS padding after every 256 data bytes (row stride 272B).
// D# per CDNA5 ISA ch.8; groups 2/3/4 NULL (2D tensor). TENSORcnt-tracked.
// ---------------------------------------------------------------------------
#if SNN_USE_TDM
__device__ __forceinline__ void snn_tdm_load_tile(const _Float16* gsrc,
                                                  unsigned lds_byte_off) {
  const unsigned long long ga = (unsigned long long)(size_t)gsrc;
  u32x4 g0;
  g0[0] = 1u;                                          // count=1, no gather
  g0[1] = lds_byte_off;                                // lds_addr
  g0[2] = (unsigned)(ga & 0xffffffffu);                // global_addr[31:0]
  g0[3] = (unsigned)((ga >> 32) & 0x01ffffffu)         // global_addr[56:32]
          | (2u << 30);                                // type = 2 (image)
  i32x8 g1;
  g1[0] = (int)((3u << 16)                             // data_size = 8 bytes
                | (1u << 20)                           // pad_enable
                | (5u << 22)                           // pad_interval: 256B
                | (3u << 25));                         // pad_amount: 4 DWORDs
  g1[1] = (int)(32u << 16);                            // tensor_dim0 = 32 units
  g1[2] = (int)(8192u << 16);                          // tensor_dim1 = 8192 rows
  g1[3] = (int)(32u << 16);                            // tile_dim0 = 32 units
  g1[4] = 16;                                          // tile_dim1 = 16 rows
  g1[5] = 32;                                          // tensor_dim0_stride = 32
  g1[6] = 0;
  g1[7] = 0;
  __builtin_amdgcn_tensor_load_to_lds(g0, g1, (i32x4)0, (i32x4)0, (i32x8)0, 0);
}
#endif

// ---------------------------------------------------------------------------
// Epilogue for one 16x16 sim tile in C layout (lane -> col N = L%16,
// VGPR v -> row M = v + 8*(L/16)).
// ---------------------------------------------------------------------------
template <bool DIAG>
__device__ __forceinline__ void snn_epilogue(const v8f& c, float scl, int tc,
                                             int dv, const int* trow,
                                             float* tot, float* pos) {
#pragma unroll
  for (int v = 0; v < 8; ++v) {
    float e = fast_exp2(c[v] * scl);         // exp(sim/T) via 2^(sim*log2e/T)
    if (DIAG) e = (v == dv) ? 0.0f : e;      // drop j==i (== subtracting diag)
    tot[v] += e;
    pos[v] += (tc == trow[v]) ? e : 0.0f;
  }
}

// ---------------------------------------------------------------------------
// Kernel 2: fused WMMA GEMM + exp epilogue. 128 threads = 4 waves; each wave
// owns 16 rows; each block covers 1/4 of the column space and writes partial
// (tot,pos) per row (unique slots -> deterministic, no atomics).
// ---------------------------------------------------------------------------
__global__ void __launch_bounds__(128) snn_main(const _Float16* __restrict__ xn,
                                                const int* __restrict__ tgt,
                                                const float* __restrict__ temp,
                                                float* __restrict__ ptot,
                                                float* __restrict__ ppos) {
  __shared__ unsigned char tcls[SNN_N];                     // class ids (8 KB)
#if SNN_USE_TDM
  __shared__ alignas(16) unsigned char btile[SNN_NBUF][16 * SNN_ROWB];
#endif

  const int tid = threadIdx.x;
  // Vectorized one-time staging of targets as bytes into LDS.
  for (int kb = tid; kb < SNN_N / 4; kb += 128) {
    const i32x4 tv = ((const i32x4*)tgt)[kb];
    unsigned packed = (unsigned)(tv[0] & 0xff) | ((unsigned)(tv[1] & 0xff) << 8) |
                      ((unsigned)(tv[2] & 0xff) << 16) |
                      ((unsigned)(tv[3] & 0xff) << 24);
    ((unsigned*)tcls)[kb] = packed;
  }
  __syncthreads();

  const int rb      = blockIdx.x / SNN_SPLIT;               // row-block id
  const int ch      = blockIdx.x % SNN_SPLIT;               // column chunk id
  const int jt0     = ch * SNN_JTPER;
  const int jtEnd   = jt0 + SNN_JTPER;

  const int wave    = tid >> 5;
  const int lane    = tid & 31;
  const int l16     = lane & 15;
  const int h       = lane >> 4;
  const int i0      = rb * 64 + wave * 16;                  // wave's 16 rows
  const int i0u     = __builtin_amdgcn_readfirstlane(i0);   // provably uniform
  const int rowbase = i0 + 8 * h;

  const float t   = fminf(fmaxf(temp[0], 0.1f), 1.0f);
  const float scl = SNN_LOG2E / t;                          // one mul per elem

  // A fragments: 4 K-chunks, register resident (32 VGPRs).
  v16h A[4];
  const _Float16* arow = xn + (size_t)(i0 + l16) * SNN_D;
#pragma unroll
  for (int kk = 0; kk < 4; ++kk) A[kk] = snn_frag(arow, kk, h);

  int trow[8];
#pragma unroll
  for (int v = 0; v < 8; ++v) trow[v] = (int)tcls[rowbase + v];

  float tot[8], pos[8];
#pragma unroll
  for (int v = 0; v < 8; ++v) { tot[v] = 0.0f; pos[v] = 0.0f; }

#if SNN_USE_TDM
  const bool issuer = (wave == 0);
  if (issuer) {                                   // prime 3 tiles in flight
#pragma unroll
    for (int p = 0; p < SNN_NBUF - 1; ++p)
      snn_tdm_load_tile(xn + (size_t)(jt0 + p) * 16 * SNN_D,
                        (unsigned)(size_t)&btile[p][0]);
  }
#endif

  for (int jt = jt0; jt < jtEnd; ++jt) {
    const int j0 = jt * 16;
#if SNN_USE_TDM
    // In-order tensor completion: <= NBUF-2 outstanding => tile jt landed.
    if (issuer) __builtin_amdgcn_s_wait_tensorcnt(SNN_NBUF - 2);
    __syncthreads();                              // publish tile jt; frees
                                                  // buffer read at jt-1
    if (issuer && jt + SNN_NBUF - 1 < jtEnd)
      snn_tdm_load_tile(xn + (size_t)(jt + SNN_NBUF - 1) * 16 * SNN_D,
                        (unsigned)(size_t)&btile[(jt + SNN_NBUF - 1) % SNN_NBUF][0]);
    const _Float16* brow =
        (const _Float16*)&btile[jt % SNN_NBUF][l16 * SNN_ROWB];
#else
    const _Float16* brow = xn + (size_t)(j0 + l16) * SNN_D;
#endif

    const int tc = (int)tcls[j0 + l16];           // column class (early issue)

    // All 8 ds_load_b128 issue before the WMMA chain (staggered waits).
    v16h B[4];
#pragma unroll
    for (int kk = 0; kk < 4; ++kk) B[kk] = snn_frag(brow, kk, h);

    v8f c = {};
#pragma unroll
    for (int kk = 0; kk < 4; ++kk)
      c = __builtin_amdgcn_wmma_f32_16x16x32_f16(
              false, A[kk], false, B[kk], (short)0, c, false, false);

    if (j0 == i0u) {                              // scalar branch: rare tile
      snn_epilogue<true>(c, scl, tc, (j0 + l16) - rowbase, trow, tot, pos);
    } else {
      snn_epilogue<false>(c, scl, tc, 0, trow, tot, pos);
    }
  }

  // Row sums live across the 16 lanes of each half -> butterfly within halves.
#pragma unroll
  for (int v = 0; v < 8; ++v) {
#pragma unroll
    for (int m = 1; m < 16; m <<= 1) {
      tot[v] += __shfl_xor(tot[v], m, 32);
      pos[v] += __shfl_xor(pos[v], m, 32);
    }
  }

  if (l16 < 8) {                                  // one row per lane
    float myT = 0.0f, myP = 0.0f;
#pragma unroll
    for (int v = 0; v < 8; ++v)
      if (l16 == v) { myT = tot[v]; myP = pos[v]; }
    const int row = rowbase + l16;
    ptot[(size_t)ch * SNN_N + row] = myT;         // unique slot per (ch,row)
    ppos[(size_t)ch * SNN_N + row] = myP;
  }
}

// ---------------------------------------------------------------------------
// Kernel 3: deterministic partial reduction, per-sample loss, per-class
// aggregation (LDS atomics) + final weighted scalar.
// ---------------------------------------------------------------------------
__global__ void __launch_bounds__(256) snn_finalize(const float* __restrict__ ptot,
                                                    const float* __restrict__ ppos,
                                                    const int* __restrict__ tgt,
                                                    float* __restrict__ out) {
  __shared__ float ssum[SNN_NCLS];
  __shared__ int   scnt[SNN_NCLS];
  const int tid = threadIdx.x;
  if (tid < SNN_NCLS) { ssum[tid] = 0.0f; scnt[tid] = 0; }
  __syncthreads();

  for (int i = tid; i < SNN_N; i += 256) {
    float tot = 0.0f, pos = 0.0f;
#pragma unroll
    for (int chv = 0; chv < SNN_SPLIT; ++chv) {   // fixed order: deterministic
      tot += ptot[(size_t)chv * SNN_N + i];
      pos += ppos[(size_t)chv * SNN_N + i];
    }
    const float lossi = (fast_log2(tot) - fast_log2(pos)) * SNN_LN2;
    const int c = tgt[i];
    atomicAdd(&scnt[c], 1);            // ds_add_u32
    atomicAdd(&ssum[c], lossi);        // ds_add_f32
  }
  __syncthreads();

  if (tid == 0) {
    float inv[SNN_NCLS];
    float invsum = 0.0f;
    for (int k = 0; k < SNN_NCLS; ++k) {
      inv[k] = (scnt[k] > 0) ? 1.0f / (float)scnt[k] : 0.0f;
      invsum += inv[k];
    }
    float acc = 0.0f, nvalid = 0.0f;
    for (int k = 0; k < SNN_NCLS; ++k) {
      if (scnt[k] > 1) {
        const float mean = ssum[k] / (float)scnt[k];
        acc += mean * (inv[k] / invsum);
        nvalid += 1.0f;
      }
    }
    out[0] = acc / nvalid;
  }
}

// ---------------------------------------------------------------------------
extern "C" void kernel_launch(void* const* d_in, const int* in_sizes, int n_in,
                              void* d_out, int out_size, void* d_ws, size_t ws_size,
                              hipStream_t stream) {
  const float* d_input = (const float*)d_in[0];   // [8192,128] f32
  const int*   d_tgt   = (const int*)d_in[1];     // [8192] int
  const float* d_temp  = (const float*)d_in[2];   // scalar
  float*       out     = (float*)d_out;

  _Float16* xn   = (_Float16*)d_ws;                                    // 2 MB
  float*    ptot = (float*)((char*)d_ws + (size_t)SNN_N * SNN_D * 2);  // 128 KB
  float*    ppos = ptot + (size_t)SNN_SPLIT * SNN_N;

  snn_normalize<<<SNN_N / 8, 256, 0, stream>>>(d_input, xn);
  snn_main<<<(SNN_N / 64) * SNN_SPLIT, 128, 0, stream>>>(xn, d_tgt, d_temp,
                                                         ptot, ppos);
  snn_finalize<<<1, 256, 0, stream>>>(ptot, ppos, d_tgt, out);
}
